// BLSTM_7430293422449
// MI455X (gfx1250) — matrix operand order repaired and checked by reading, not compiled, not run on hardware
//
#include <hip/hip_runtime.h>
#include <hip/hip_bf16.h>

// ---------------------------------------------------------------------------
// BiLSTM LM for MI455X (gfx1250, wave32, WMMA).
//   B=8, T=256, E=512, H=1024, V=32000
// Plan:
//   1) embed+cast -> xe2 [2048=t*8+b][512] bf16
//   2) convert+swizzle W_f/W_b/fc_W/fcf_W into WMMA B-tile layout (bf16),
//      LDS-tiled so both global reads and writes are fully coalesced
//   3) zx[dir] = xe2 @ W[dir] + b[dir]    (bf16 WMMA GEMM, f32 out)
//   4) persistent LSTM kernel: 64 WGs, U slices resident in LDS (256KB/WG),
//      1 grid barrier per timestep, double-buffered bf16 h state,
//      writes hcat [2048 rows=b*256+t][2048 cols=dir*1024+h] bf16
//   5) out1 = hcat @ fc_W + fc_b ; out2 = hcat[:, :1024] @ fcf_W + fcf_b
// ---------------------------------------------------------------------------

typedef __attribute__((ext_vector_type(16))) __bf16 v16bf;
typedef __attribute__((ext_vector_type(8)))  float  v8f;

union V16 { uint4 q[2]; v16bf v; };

__device__ inline v8f wmma_bf16(v16bf a, v16bf b, v8f c) {
  return __builtin_amdgcn_wmma_f32_16x16x32_bf16(false, a, false, b, (short)0, c,
                                                 false, false);
}

__device__ inline __bf16 f2bf(float f) {
  unsigned u = __builtin_bit_cast(unsigned, f);
  unsigned r = (u + 0x7FFFu + ((u >> 16) & 1u)) >> 16;   // round-to-nearest-even
  unsigned short s = (unsigned short)r;
  return __builtin_bit_cast(__bf16, s);
}

__device__ inline float sigmoidf_(float x) { return 1.0f / (1.0f + __expf(-x)); }

// ---------------------------------------------------------------------------
// 1) Embedding lookup + f32->bf16 cast. xe2 row r = t*8+b.
// ---------------------------------------------------------------------------
__global__ void embed_kernel(const int* __restrict__ x, const float* __restrict__ emb,
                             __bf16* __restrict__ xe2) {
  int r = blockIdx.x;                 // 0..2047
  int t = r >> 3, b = r & 7;
  int tok = x[b * 256 + t];
  const float* src = emb + (long)tok * 512;
  __bf16* dst = xe2 + (long)r * 512;
  for (int e = threadIdx.x; e < 512; e += blockDim.x) dst[e] = f2bf(src[e]);
}

// ---------------------------------------------------------------------------
// 2) Convert f32 [K][N] weight to bf16, pre-swizzled into WMMA B-operand tiles:
//    dst[(ntile*(K/32)+kb)*512 + lane*16 + e], lane=((k&31)>>4)*16|(n&15), e=k&15
//    LDS-tiled: coalesced 64-float row reads; each thread writes 16 contiguous
//    bytes of the swizzled layout (merges to one b128 store).
//    Grid: (N/64, K/32), 256 threads.
// ---------------------------------------------------------------------------
__global__ __launch_bounds__(256) void convswz_kernel(
    const float* __restrict__ src, __bf16* __restrict__ dst, int K, int N) {
  __shared__ float tile[32][64];
  const int nt4 = blockIdx.x;          // 64-column block
  const int kb  = blockIdx.y;          // 32-row K block
  const int k5  = K >> 5;
  const int tid = threadIdx.x;
#pragma unroll
  for (int i = 0; i < 8; ++i) {
    int flat = i * 256 + tid;
    int kloc = flat >> 6, nloc = flat & 63;
    tile[kloc][nloc] = src[(long)(kb * 32 + kloc) * N + nt4 * 64 + nloc];
  }
  __syncthreads();
#pragma unroll
  for (int i = 0; i < 8; ++i) {
    int flat = tid * 8 + i;            // 8 consecutive dst elements per thread
    int e    = flat & 15;
    int lane = (flat >> 4) & 31;
    int ntl  = flat >> 9;              // 0..3 local ntile
    int kloc = ((lane >> 4) << 4) + e;
    int nloc = ntl * 16 + (lane & 15);
    dst[((long)(nt4 * 4 + ntl) * k5 + kb) * 512 + lane * 16 + e] =
        f2bf(tile[kloc][nloc]);
  }
}

// ---------------------------------------------------------------------------
// 3/5) bf16 WMMA GEMM: C[M][N](f32) = A[M][K](bf16,row-major,lda) @ Bswz + bias
//      WG = 256 thr (8 waves); macro-tile 128M x 64N; wave = 2x2 16x16 tiles.
//      M multiple of 128, N multiple of 64, K multiple of 32 (all hold here).
// ---------------------------------------------------------------------------
__global__ __launch_bounds__(256) void wmma_gemm_kernel(
    const __bf16* __restrict__ A, int lda, const __bf16* __restrict__ Bs,
    const float* __restrict__ bias, float* __restrict__ C, int ldc, int K) {
  const int lane = threadIdx.x & 31;
  const int w    = threadIdx.x >> 5;
  const int mq = w & 3, nh = w >> 2;
  const int mt0 = blockIdx.y * 8 + mq * 2;
  const int nt0 = blockIdx.x * 4 + nh * 2;
  const int k5 = K >> 5;
  const int mrow = lane & 15, kh = lane >> 4;

  const __bf16* a0 = A + (long)(mt0 * 16 + mrow) * lda + kh * 16;
  const __bf16* a1 = a0 + (long)16 * lda;
  const __bf16* b0 = Bs + (long)nt0 * k5 * 512 + lane * 16;
  const __bf16* b1 = b0 + (long)k5 * 512;

  v8f acc00 = {}, acc01 = {}, acc10 = {}, acc11 = {};
  for (int kb = 0; kb < k5; ++kb) {
    V16 av0, av1, bv0, bv1;
    av0.q[0] = *(const uint4*)(a0);      av0.q[1] = *(const uint4*)(a0 + 8);
    av1.q[0] = *(const uint4*)(a1);      av1.q[1] = *(const uint4*)(a1 + 8);
    bv0.q[0] = ((const uint4*)b0)[0];    bv0.q[1] = ((const uint4*)b0)[1];
    bv1.q[0] = ((const uint4*)b1)[0];    bv1.q[1] = ((const uint4*)b1)[1];
    acc00 = wmma_bf16(av0.v, bv0.v, acc00);
    acc01 = wmma_bf16(av0.v, bv1.v, acc01);
    acc10 = wmma_bf16(av1.v, bv0.v, acc10);
    acc11 = wmma_bf16(av1.v, bv1.v, acc11);
    a0 += 32; a1 += 32; b0 += 512; b1 += 512;
  }

  v8f* accs[2][2] = {{&acc00, &acc01}, {&acc10, &acc11}};
  for (int i = 0; i < 2; ++i)
    for (int j = 0; j < 2; ++j) {
      int n = (nt0 + j) * 16 + (lane & 15);
      float bn = bias[n];
      int mbase = (mt0 + i) * 16 + kh * 8;
      v8f a = *accs[i][j];
#pragma unroll
      for (int r = 0; r < 8; ++r) C[(long)(mbase + r) * ldc + n] = a[r] + bn;
    }
}

// ---------------------------------------------------------------------------
// 4) Persistent bidirectional LSTM. 64 WGs x 256 thr; WG owns 2 (dir,j) column
//    blocks; U slice lives in LDS (2 x 128KB, bf16, B-tile layout).
//    One grid barrier per step; h double-buffered in global bf16.
// ---------------------------------------------------------------------------
__device__ inline void grid_barrier(unsigned* cnt, unsigned target) {
  __syncthreads();
  if (threadIdx.x == 0) {
    __threadfence();
    __hip_atomic_fetch_add(cnt, 1u, __ATOMIC_RELEASE, __HIP_MEMORY_SCOPE_AGENT);
    while (__hip_atomic_load(cnt, __ATOMIC_ACQUIRE, __HIP_MEMORY_SCOPE_AGENT) < target) {
      __builtin_amdgcn_s_sleep(8);
    }
  }
  __syncthreads();
}

__global__ __launch_bounds__(256) void lstm_kernel(
    const float* __restrict__ U_f, const float* __restrict__ U_b,
    const float* __restrict__ zx,        // [2][2048][4096] f32 (xe@W + b)
    __bf16* __restrict__ hbf,            // [2 dir][2 buf][16][1024] bf16
    __bf16* __restrict__ hcat,           // [2048][2048] bf16
    unsigned* __restrict__ barcnt) {
  __shared__ __attribute__((aligned(16))) __bf16 Us[2][4 * 32 * 512];  // 256 KB
  __shared__ float red[8][32][8];                                      // 8 KB
  __shared__ float zbuf[4][16][16];                                    // 4 KB
  __shared__ float cst[2][8][16];                                      // 1 KB

  const int tid = threadIdx.x;
  const int lane = tid & 31, w = tid >> 5;
  const int nblk = gridDim.x;
  unsigned barseq = 0;

  // zero c-state (256 floats == 256 threads)
  ((float*)cst)[tid] = 0.0f;

  // zero h state buffers (blocks 0..3 cover 4*16384 bf16)
  if (blockIdx.x < 4) {
    for (int i = tid; i < 16384; i += 256) hbf[blockIdx.x * 16384 + i] = f2bf(0.0f);
  }

  // preload U slices into LDS (f32 -> bf16, B-tile swizzle)
  for (int ci = 0; ci < 2; ++ci) {
    int c = blockIdx.x * 2 + ci;
    int dir = c >> 6, j = c & 63;
    const float* U = dir ? U_b : U_f;
    int kk = tid >> 6, g = (tid >> 4) & 3, n = tid & 15;
    for (int k0 = 0; k0 < 1024; k0 += 4) {
      int k = k0 + kk;
      float v = U[(long)k * 4096 + g * 1024 + j * 16 + n];
      int kb = k >> 5, kr = k & 31;
      int lane2 = ((kr >> 4) << 4) | n;
      int e = kr & 15;
      Us[ci][(g * 32 + kb) * 512 + lane2 * 16 + e] = f2bf(v);
    }
  }
  __syncthreads();
  grid_barrier(barcnt, (++barseq) * nblk);   // h zeroed + LDS ready everywhere

  const int g = w & 3, kh2 = w >> 2;         // wave roles: gate, K-half
  const int ln = lane & 15, kh = lane >> 4;

  for (int s = 0; s < 256; ++s) {
    const int rbuf = s & 1, wbuf = rbuf ^ 1;
    for (int ci = 0; ci < 2; ++ci) {
      const int c = blockIdx.x * 2 + ci;
      const int dir = c >> 6, j = c & 63;
      const int t = dir ? (255 - s) : s;

      // ---- z_partial = h @ U  (wave: one gate tile, half of K) ----
      const __bf16* hrow = hbf + ((dir * 2 + rbuf) * 16 + ln) * 1024;
      const __bf16* bp = &Us[ci][(g * 32 + kh2 * 16) * 512 + lane * 16];
      const int kbase = kh2 * 512 + kh * 16;
      v8f acc = {};
#pragma unroll 4
      for (int kb = 0; kb < 16; ++kb) {
        V16 a, b;
        const __bf16* ap = hrow + kbase + kb * 32;
        a.q[0] = *(const uint4*)(ap);
        a.q[1] = *(const uint4*)(ap + 8);
        b.q[0] = ((const uint4*)bp)[0];
        b.q[1] = ((const uint4*)bp)[1];
        acc = wmma_bf16(a.v, b.v, acc);
        bp += 512;
      }
#pragma unroll
      for (int r = 0; r < 8; ++r) red[w][lane][r] = acc[r];
      __syncthreads();

      // ---- reduce K-halves, add zx (xe@W+b), stage gates ----
      if (w < 4) {
        int col = w * 1024 + j * 16 + ln;
        const float* zxp = zx + ((long)dir * 2048 + t * 8) * 4096 + col;
#pragma unroll
        for (int r = 0; r < 8; ++r) {
          int mm = kh * 8 + r;
          float zz = red[w][lane][r] + red[w + 4][lane][r];
          if (mm < 8) zz += zxp[mm * 4096];
          zbuf[w][mm][ln] = zz;
        }
      }
      __syncthreads();

      // ---- elementwise gate math, c/h update, publish h ----
      if (tid < 128) {
        int mm = tid >> 4, n = tid & 15;
        float fg = sigmoidf_(zbuf[0][mm][n]);
        float ig = sigmoidf_(zbuf[1][mm][n]);
        float gg = tanhf(zbuf[2][mm][n]);
        float og = sigmoidf_(zbuf[3][mm][n]);
        float c2 = fg * cst[ci][mm][n] + ig * gg;
        cst[ci][mm][n] = c2;
        float h = og * tanhf(c2);
        __bf16 hb = f2bf(h);
        hbf[((dir * 2 + wbuf) * 16 + mm) * 1024 + j * 16 + n] = hb;
        hcat[(long)(mm * 256 + t) * 2048 + dir * 1024 + j * 16 + n] = hb;
      }
      __syncthreads();
    }
    grid_barrier(barcnt, (++barseq) * nblk);
  }
}

// ---------------------------------------------------------------------------
// Host launcher
// ---------------------------------------------------------------------------
extern "C" void kernel_launch(void* const* d_in, const int* in_sizes, int n_in,
                              void* d_out, int out_size, void* d_ws, size_t ws_size,
                              hipStream_t stream) {
  (void)in_sizes; (void)n_in; (void)out_size; (void)ws_size;
  const int*   x     = (const int*)d_in[0];
  const float* emb   = (const float*)d_in[1];
  const float* W_f   = (const float*)d_in[2];
  const float* U_f   = (const float*)d_in[3];
  const float* b_f   = (const float*)d_in[4];
  const float* W_b   = (const float*)d_in[5];
  const float* U_b   = (const float*)d_in[6];
  const float* b_b   = (const float*)d_in[7];
  const float* fc_W  = (const float*)d_in[8];
  const float* fc_b  = (const float*)d_in[9];
  const float* fcf_W = (const float*)d_in[10];
  const float* fcf_b = (const float*)d_in[11];
  float* out = (float*)d_out;

  char* ws = (char*)d_ws;
  size_t off = 0;
  auto alloc = [&](size_t bytes) -> void* {
    void* p = ws + off;
    off += (bytes + 255) & ~(size_t)255;
    return p;
  };
  unsigned* bar  = (unsigned*)alloc(256);
  __bf16* hbf    = (__bf16*)alloc((size_t)2 * 2 * 16 * 1024 * 2);
  __bf16* xe2    = (__bf16*)alloc((size_t)2048 * 512 * 2);
  __bf16* Ws0    = (__bf16*)alloc((size_t)512 * 4096 * 2);
  __bf16* Ws1    = (__bf16*)alloc((size_t)512 * 4096 * 2);
  float*  zx     = (float*)alloc((size_t)2 * 2048 * 4096 * 4);
  __bf16* hcat   = (__bf16*)alloc((size_t)2048 * 2048 * 2);
  __bf16* fcWs   = (__bf16*)alloc((size_t)2048 * 32000 * 2);
  __bf16* fcfWs  = (__bf16*)alloc((size_t)1024 * 32000 * 2);

  hipMemsetAsync(bar, 0, 256, stream);

  embed_kernel<<<2048, 256, 0, stream>>>(x, emb, xe2);

  // swizzled bf16 weight images (grid = (N/64, K/32))
  convswz_kernel<<<dim3(64, 16),  256, 0, stream>>>(W_f,   Ws0,   512,  4096);
  convswz_kernel<<<dim3(64, 16),  256, 0, stream>>>(W_b,   Ws1,   512,  4096);
  convswz_kernel<<<dim3(500, 64), 256, 0, stream>>>(fc_W,  fcWs,  2048, 32000);
  convswz_kernel<<<dim3(500, 32), 256, 0, stream>>>(fcf_W, fcfWs, 1024, 32000);

  // zx[dir] = xe2 @ W[dir] + b[dir]   (M=2048, N=4096, K=512)
  wmma_gemm_kernel<<<dim3(64, 16), 256, 0, stream>>>(xe2, 512, Ws0, b_f, zx, 4096, 512);
  wmma_gemm_kernel<<<dim3(64, 16), 256, 0, stream>>>(xe2, 512, Ws1, b_b,
                                                     zx + (size_t)2048 * 4096, 4096, 512);

  lstm_kernel<<<64, 256, 0, stream>>>(U_f, U_b, zx, hbf, hcat, bar);

  // out1 = hcat @ fc_W + fc_b   (M=2048, N=32000, K=2048)
  wmma_gemm_kernel<<<dim3(500, 16), 256, 0, stream>>>(hcat, 2048, fcWs, fc_b, out,
                                                      32000, 2048);
  // out2 = hcat[:, :1024] @ fcf_W + fcf_b   (K=1024)
  wmma_gemm_kernel<<<dim3(500, 16), 256, 0, stream>>>(hcat, 2048, fcfWs, fcf_b,
                                                      out + (size_t)2048 * 32000,
                                                      32000, 1024);
}
